// GNNModel_51419348468094
// MI455X (gfx1250) — compile-verified, matrix operand order checked by reading, not run on hardware
//
#include <hip/hip_runtime.h>
#include <hip/hip_bf16.h>

typedef __attribute__((ext_vector_type(16))) _Float16 v16h;
typedef __attribute__((ext_vector_type(8)))  float    v8f;

#define NN  100000   // nodes
#define NE  1600000  // edges
#define NG  256      // graphs
#define NC  10       // classes

// ---------------- utility ----------------
__global__ void fill_kernel(float* __restrict__ p, long n, float v) {
  long i = (long)blockIdx.x * blockDim.x + threadIdx.x;
  long stride = (long)gridDim.x * blockDim.x;
  for (; i < n; i += stride) p[i] = v;
}

// deg[dst] += 1 per edge (deg pre-initialized to 1.0 for self-loops)
__global__ void degree_kernel(const int* __restrict__ dst, float* __restrict__ deg, int E) {
  int i = blockIdx.x * blockDim.x + threadIdx.x;
  if (i < E) atomicAdd(&deg[dst[i]], 1.0f);
}

__global__ void rsqrt_kernel(float* __restrict__ d, int n) {
  int i = blockIdx.x * blockDim.x + threadIdx.x;
  if (i < n) d[i] = rsqrtf(d[i]);  // deg >= 1 always (self loop)
}

__global__ void norm_kernel(const int* __restrict__ src, const int* __restrict__ dst,
                            const float* __restrict__ dinv, float* __restrict__ nrm, int E) {
  int i = blockIdx.x * blockDim.x + threadIdx.x;
  if (i < E) nrm[i] = dinv[src[i]] * dinv[dst[i]];
}

// ---------------- layer 1: [N,3] @ [3,32] (too skinny for WMMA) --------------
__global__ void gemm3_kernel(const float* __restrict__ x, const float* __restrict__ W,
                             float* __restrict__ out, int n) {
  int i = blockIdx.x * blockDim.x + threadIdx.x;
  if (i >= n * 32) return;
  int node = i >> 5, f = i & 31;
  const float* xr = x + (long)node * 3;
  out[i] = xr[0] * W[f] + xr[1] * W[32 + f] + xr[2] * W[64 + f];
}

// ---------------- generic WMMA GEMM: D[M,N] = A[M,K] @ B[K,N] (+bias,relu) ---
// M % 16 == 0, N % 16 == 0, K compile-time (32/64/128) -> fully unrolled
// K-chain of v_wmma. One wave computes one 16x16 tile; blockDim = (32, 4).
template <int K, bool FUSE>
__global__ __launch_bounds__(128)
void gemm_wmma_kernel(const float* __restrict__ A, const float* __restrict__ B,
                      float* __restrict__ D, const float* __restrict__ bias,
                      int N) {
  const int lane    = threadIdx.x;                 // 0..31 (wave32)
  const int rowBase = blockIdx.x * 16;
  const int colBase = (blockIdx.y * blockDim.y + threadIdx.y) * 16;
  if (colBase >= N) return;                        // wave-uniform guard

  const int halfSel = lane >> 4;                   // 0: K 0..15, 1: K 16..31 of the chunk
  const int l15     = lane & 15;
  const int rowA    = rowBase + l15;               // A row held by this lane
  const int colB    = colBase + l15;               // B/C/D column held by this lane

  // hoist bias: one scalar per lane (per output column), no branch in epilogue
  const float bv = FUSE ? bias[colB] : 0.0f;

  v8f c = {};
#pragma unroll
  for (int kb = 0; kb < K; kb += 32) {
    // A 16x32 f16 tile: lane holds 16 consecutive K values of its row.
    // 16B-aligned: buffers are 256B aligned, row*K*4 is a multiple of 128B.
    const float4* ap = (const float4*)(A + (long)rowA * K + kb + halfSel * 16);
    v16h a;
#pragma unroll
    for (int j = 0; j < 4; ++j) {
      float4 q = ap[j];
      a[4 * j + 0] = (_Float16)q.x;
      a[4 * j + 1] = (_Float16)q.y;
      a[4 * j + 2] = (_Float16)q.z;
      a[4 * j + 3] = (_Float16)q.w;
    }

    // B 32x16 f16 tile: lane holds its column; VGPR j holds K = base+2j, base+2j+1
    v16h b;
#pragma unroll
    for (int j = 0; j < 8; ++j) {
      int k0 = kb + halfSel * 16 + 2 * j;
      b[2 * j]     = (_Float16)B[(long)k0 * N + colB];
      b[2 * j + 1] = (_Float16)B[(long)(k0 + 1) * N + colB];
    }

    c = __builtin_amdgcn_wmma_f32_16x16x32_f16(
        /*neg_a=*/false, a, /*neg_b=*/false, b,
        /*c_mod=*/(short)0, c, /*reuse_a=*/false, /*reuse_b=*/false);
  }

  // C/D layout: VGPR r -> M = r (lanes 0-15) or r+8 (lanes 16-31); N = l15
#pragma unroll
  for (int r = 0; r < 8; ++r) {
    int row = rowBase + r + halfSel * 8;
    float v = c[r];
    if (FUSE) v = fmaxf(v + bv, 0.0f);
    D[(long)row * N + colB] = v;
  }
}

// ---------------- edge scatter: out[dst] += msg[src] * norm ------------------
// blockDim = (F, 256/F): one edge per y-slice, features across x (coalesced).
__global__ void scatter_kernel(const int* __restrict__ src, const int* __restrict__ dst,
                               const float* __restrict__ nrm, const float* __restrict__ msg,
                               float* __restrict__ out, int E, int F) {
  int e = blockIdx.x * blockDim.y + threadIdx.y;
  if (e >= E) return;
  int f = threadIdx.x;
  float v = msg[(long)src[e] * F + f] * nrm[e];
  atomicAdd(&out[(long)dst[e] * F + f], v);
}

// ---------------- finalize: out = relu(out + msg*dinv^2 + b) -----------------
__global__ void finalize_kernel(const float* __restrict__ msg, const float* __restrict__ dinv,
                                const float* __restrict__ bias, float* __restrict__ out,
                                long total, int lf /* log2(F) */) {
  long i = (long)blockIdx.x * blockDim.x + threadIdx.x;
  long stride = (long)gridDim.x * blockDim.x;
  int F = 1 << lf;
  for (; i < total; i += stride) {
    int node = (int)(i >> lf);
    int f = (int)(i & (F - 1));
    float di = dinv[node];
    float v = out[i] + msg[i] * di * di + bias[f];
    out[i] = fmaxf(v, 0.0f);
  }
}

// ---------------- global mean pool (F = 128) ---------------------------------
__global__ void pool_kernel(const float* __restrict__ h, const int* __restrict__ batch,
                            float* __restrict__ pooled, float* __restrict__ cnt, int n) {
  int node = blockIdx.x * blockDim.y + threadIdx.y;
  if (node >= n) return;
  int g = batch[node];
  int f = threadIdx.x;
  atomicAdd(&pooled[(long)g * 128 + f], h[(long)node * 128 + f]);
  if (f == 0) atomicAdd(&cnt[g], 1.0f);
}

__global__ void pooldiv_kernel(float* __restrict__ pooled, const float* __restrict__ cnt) {
  int i = blockIdx.x * blockDim.x + threadIdx.x;
  if (i < NG * 128) pooled[i] /= fmaxf(cnt[i >> 7], 1.0f);
}

// ---------------- FC head: [256,64] @ [64,10] + b ----------------------------
__global__ void fc2_kernel(const float* __restrict__ in, const float* __restrict__ W,
                           const float* __restrict__ b, float* __restrict__ out) {
  int i = blockIdx.x * blockDim.x + threadIdx.x;
  if (i >= NG * NC) return;
  int r = i / NC, c = i % NC;
  float acc = b[c];
#pragma unroll
  for (int k = 0; k < 64; ++k) acc += in[r * 64 + k] * W[k * NC + c];
  out[i] = acc;
}

extern "C" void kernel_launch(void* const* d_in, const int* in_sizes, int n_in,
                              void* d_out, int out_size, void* d_ws, size_t ws_size,
                              hipStream_t stream) {
  (void)in_sizes; (void)n_in; (void)out_size; (void)ws_size;

  const float* x     = (const float*)d_in[0];
  const int*   eidx  = (const int*)d_in[1];   // [2, NE] flat
  const int*   batch = (const int*)d_in[2];
  const float* W1  = (const float*)d_in[3];
  const float* b1  = (const float*)d_in[4];
  const float* W2  = (const float*)d_in[5];
  const float* b2  = (const float*)d_in[6];
  const float* W3  = (const float*)d_in[7];
  const float* b3  = (const float*)d_in[8];
  const float* fW1 = (const float*)d_in[9];
  const float* fb1 = (const float*)d_in[10];
  const float* fW2 = (const float*)d_in[11];
  const float* fb2 = (const float*)d_in[12];

  const int* src = eidx;
  const int* dst = eidx + NE;

  // ---- workspace carve-up (256B aligned) ----
  char* wsp = (char*)d_ws;
  auto alloc = [&](size_t nfloats) -> float* {
    float* p = (float*)wsp;
    wsp += ((nfloats * sizeof(float) + 255) / 256) * 256;
    return p;
  };
  float* dinv   = alloc(NN);                 // degrees -> dinv (in place)
  float* nrm    = alloc(NE);                 // per-edge norm (reused all 3 layers)
  float* bufA   = alloc((size_t)NN * 128);   // tmp = h @ W
  float* bufB   = alloc((size_t)NN * 128);   // h (aggregated)
  float* pooled = alloc((size_t)NG * 128);
  float* cnt    = alloc(NG);
  float* fc1o   = alloc((size_t)NG * 64);

  // ---- degree normalization (computed once; shared across layers) ----
  fill_kernel<<<512, 256, 0, stream>>>(dinv, NN, 1.0f);                       // self loops
  degree_kernel<<<(NE + 255) / 256, 256, 0, stream>>>(dst, dinv, NE);
  rsqrt_kernel<<<(NN + 255) / 256, 256, 0, stream>>>(dinv, NN);
  norm_kernel<<<(NE + 255) / 256, 256, 0, stream>>>(src, dst, dinv, nrm, NE);

  // ---- layer 1: 3 -> 32 ----
  gemm3_kernel<<<(NN * 32 + 255) / 256, 256, 0, stream>>>(x, W1, bufA, NN);
  fill_kernel<<<2048, 256, 0, stream>>>(bufB, (long)NN * 32, 0.0f);
  { dim3 blk(32, 8); scatter_kernel<<<(NE + 7) / 8, blk, 0, stream>>>(src, dst, nrm, bufA, bufB, NE, 32); }
  finalize_kernel<<<4096, 256, 0, stream>>>(bufA, dinv, b1, bufB, (long)NN * 32, 5);

  // ---- layer 2: 32 -> 64 (WMMA, K=32) ----
  { dim3 blk(32, 4); dim3 grd(NN / 16, 1);
    gemm_wmma_kernel<32, false><<<grd, blk, 0, stream>>>(bufB, W2, bufA, nullptr, 64); }
  fill_kernel<<<2048, 256, 0, stream>>>(bufB, (long)NN * 64, 0.0f);
  { dim3 blk(64, 4); scatter_kernel<<<(NE + 3) / 4, blk, 0, stream>>>(src, dst, nrm, bufA, bufB, NE, 64); }
  finalize_kernel<<<4096, 256, 0, stream>>>(bufA, dinv, b2, bufB, (long)NN * 64, 6);

  // ---- layer 3: 64 -> 128 (WMMA, K=64) ----
  { dim3 blk(32, 4); dim3 grd(NN / 16, 2);
    gemm_wmma_kernel<64, false><<<grd, blk, 0, stream>>>(bufB, W3, bufA, nullptr, 128); }
  fill_kernel<<<4096, 256, 0, stream>>>(bufB, (long)NN * 128, 0.0f);
  { dim3 blk(128, 2); scatter_kernel<<<(NE + 1) / 2, blk, 0, stream>>>(src, dst, nrm, bufA, bufB, NE, 128); }
  finalize_kernel<<<4096, 256, 0, stream>>>(bufA, dinv, b3, bufB, (long)NN * 128, 7);

  // ---- global mean pool ----
  fill_kernel<<<64, 256, 0, stream>>>(pooled, (long)NG * 128, 0.0f);
  fill_kernel<<<1, 256, 0, stream>>>(cnt, NG, 0.0f);
  { dim3 blk(128, 2); pool_kernel<<<(NN + 1) / 2, blk, 0, stream>>>(bufB, batch, pooled, cnt, NN); }
  pooldiv_kernel<<<(NG * 128 + 255) / 256, 256, 0, stream>>>(pooled, cnt);

  // ---- FC1: [256,128] @ [128,64] + bias, relu (WMMA, K=128, fused) ----
  { dim3 blk(32, 4); dim3 grd(NG / 16, 1);
    gemm_wmma_kernel<128, true><<<grd, blk, 0, stream>>>(pooled, fW1, fc1o, fb1, 64); }

  // ---- FC2: [256,64] @ [64,10] + bias -> d_out ----
  fc2_kernel<<<(NG * NC + 255) / 256, 256, 0, stream>>>(fc1o, fW2, fb2, (float*)d_out);
}
